// MoEEncoderLayer_78365973283406
// MI455X (gfx1250) — matrix operand order repaired and checked by reading, not compiled, not run on hardware
//
#include <hip/hip_runtime.h>
#include <hip/hip_bf16.h>
#include <math.h>

#define DEV __device__ __forceinline__

typedef __bf16 bf16_t;
typedef __attribute__((ext_vector_type(16))) __bf16 bf16x16;
typedef __attribute__((ext_vector_type(8)))  float  f32x8;

// ---- problem dims -----------------------------------------------------------
constexpr int Bb = 16, Cc = 1024, Dd = 512, Hh = 8;
constexpr int Ee = 8, FFf = 2048;
constexpr int Tt = Bb * Cc;  // 16384
constexpr int CAPc = 5120;
constexpr float EPSf = 1e-5f;

// ---- fragment access --------------------------------------------------------
// All LDS tiles are stored pre-swizzled: fragment layout is [tile][lane][16],
// i.e. each lane's 16 bf16 operand elements are 32 contiguous bytes.
//  A (16x32 MxK): lane = (m&15) + 16*((k>>3)&1); pos = 8*(k>>4) + (k&7)
//  B (32x16 KxN): lane = (n&15) + 16*((k>>3)&1); pos = 8*(k>>4) + (k&7)
// A fragment load is exactly two ds_load_b128.
DEV bf16x16 frag_ld(const bf16_t* p) {
  const int lane = threadIdx.x & 31;
  union { uint4 u[2]; bf16x16 v; } x;
  const uint4* q = (const uint4*)(p + lane * 16);
  x.u[0] = q[0];
  x.u[1] = q[1];
  return x.v;
}
DEV f32x8 wmma_bf16(bf16x16 a, bf16x16 b, f32x8 c) {
  return __builtin_amdgcn_wmma_f32_16x16x32_bf16(false, a, false, b, (short)0, c,
                                                 false, false);
}
DEV void cp16B(bf16_t* dst, const bf16_t* src) {
  *(uint4*)dst = *(const uint4*)src;
}
DEV void store8_bf16(bf16_t* dst, const float* v) {
  union { bf16_t h[8]; uint4 u; } w;
#pragma unroll
  for (int j = 0; j < 8; ++j) w.h[j] = (bf16_t)v[j];
  *(uint4*)dst = w.u;
}
// C/D (16x16 f32): reg r -> row = 8*(lane>>4)+r, col = lane&15

// ---- tiled transpose + fp32->bf16 convert: out[b][c][r] = in[b][r][c] -------
__global__ __launch_bounds__(256) void transpose_f2b(const float* __restrict__ in,
                                                     bf16_t* __restrict__ out,
                                                     int R, int C) {
  __shared__ float t[32][33];
  const int c0 = blockIdx.x * 32, r0 = blockIdx.y * 32;
  const size_t bo = (size_t)blockIdx.z * R * C;
  const int tid = threadIdx.x, cl = tid & 31, rl = tid >> 5;
#pragma unroll
  for (int i = 0; i < 4; ++i)
    t[rl + i * 8][cl] = in[bo + (size_t)(r0 + rl + i * 8) * C + c0 + cl];
  __syncthreads();
#pragma unroll
  for (int i = 0; i < 4; ++i) {
    int oc = rl + i * 8, orr = cl;
    out[bo + (size_t)(c0 + oc) * R + r0 + orr] = (bf16_t)t[orr][oc];
  }
}

// ---- generic tiled WMMA GEMM: Y = A @ Wt^T + bias ---------------------------
// Wt is bf16, pre-transposed [N][K]. Block tile 128x64, BK=32, 256 threads.
// MODE: 0 = f32 out [M][N]; 1 = bf16 out head-split [bh][c][hd];
//       2 = bf16 out head-split transposed [bh][hd][c]
template <bool A_BF16, int MODE>
__global__ __launch_bounds__(256) void gemm_wmma(const void* __restrict__ Ap,
                                                 const bf16_t* __restrict__ Wt,
                                                 const float* __restrict__ bias,
                                                 void* __restrict__ Out, int M,
                                                 int N, int Kd) {
  __shared__ alignas(16) bf16_t Aswz[8 * 512];  // [mt8][lane][16]
  __shared__ alignas(16) bf16_t Bswz[4 * 512];  // [nt4][lane][16]
  const int tid = threadIdx.x, lane = tid & 31, wid = tid >> 5;
  const int mw = wid & 3, nw = wid >> 2;
  const int m0 = blockIdx.x * 128, n0 = blockIdx.y * 64;
  f32x8 acc[2][2] = {};
  for (int kb = 0; kb < Kd; kb += 32) {
    // --- stage A (swizzled) ---
#pragma unroll
    for (int i = 0; i < 2; ++i) {
      int G = i * 256 + tid;
      int r = G >> 2, k = (G & 3) * 8;
      bf16_t* dst = Aswz + (r >> 4) * 512 +
                    (((r & 15) + 16 * ((k >> 3) & 1)) << 4) + 8 * (k >> 4);
      if constexpr (A_BF16) {
        cp16B(dst, (const bf16_t*)Ap + (size_t)(m0 + r) * Kd + kb + k);
      } else {
        const float* A = (const float*)Ap + (size_t)(m0 + r) * Kd + kb + k;
        float4 v0 = *(const float4*)A;
        float4 v1 = *(const float4*)(A + 4);
        float vv[8] = {v0.x, v0.y, v0.z, v0.w, v1.x, v1.y, v1.z, v1.w};
        store8_bf16(dst, vv);
      }
    }
    // --- stage B (swizzled, from transposed bf16 weights: contiguous in K) ---
    {
      int n = tid >> 2, k = (tid & 3) * 8;
      bf16_t* dst = Bswz + (n >> 4) * 512 +
                    (((n & 15) + 16 * ((k >> 3) & 1)) << 4) + 8 * (k >> 4);
      cp16B(dst, Wt + (size_t)(n0 + n) * Kd + kb + k);
    }
    __syncthreads();
    bf16x16 a0 = frag_ld(Aswz + (2 * mw) * 512);
    bf16x16 a1 = frag_ld(Aswz + (2 * mw + 1) * 512);
    bf16x16 b0 = frag_ld(Bswz + (2 * nw) * 512);
    bf16x16 b1 = frag_ld(Bswz + (2 * nw + 1) * 512);
    acc[0][0] = wmma_bf16(a0, b0, acc[0][0]);
    acc[0][1] = wmma_bf16(a0, b1, acc[0][1]);
    acc[1][0] = wmma_bf16(a1, b0, acc[1][0]);
    acc[1][1] = wmma_bf16(a1, b1, acc[1][1]);
    __syncthreads();
  }
  const int g = lane >> 4, n = lane & 15;
#pragma unroll
  for (int i = 0; i < 2; ++i)
#pragma unroll
    for (int j = 0; j < 2; ++j)
#pragma unroll
      for (int r = 0; r < 8; ++r) {
        int row = m0 + 32 * mw + 16 * i + 8 * g + r;
        int col = n0 + 32 * nw + 16 * j + n;
        float v = acc[i][j][r] + bias[col];
        if constexpr (MODE == 0) {
          ((float*)Out)[(size_t)row * N + col] = v;
        } else {
          int b = row >> 10, c = row & 1023;
          int h = col >> 6, d = col & 63;
          size_t idx;
          if constexpr (MODE == 1)
            idx = (((size_t)(b * Hh + h)) << 16) + ((size_t)c << 6) + d;
          else
            idx = (((size_t)(b * Hh + h)) << 16) + ((size_t)d << 10) + c;
          ((bf16_t*)Out)[idx] = (bf16_t)v;
        }
      }
}

// ---- flash attention: grid (B*H, C/64), 256 threads -------------------------
// Q,K in [bh][c][hd]; V in transposed [bh][hd][c]
__global__ __launch_bounds__(256) void flash_attn(const bf16_t* __restrict__ Q,
                                                  const bf16_t* __restrict__ Kb,
                                                  const bf16_t* __restrict__ Vt,
                                                  const int* __restrict__ mask,
                                                  bf16_t* __restrict__ ctx) {
  __shared__ alignas(16) bf16_t Qs[8 * 512];  // [ck2][mt4][lane][16]
  __shared__ alignas(16) bf16_t Ks[4 * 512];  // [ck2][nt2][lane][16]
  __shared__ alignas(16) bf16_t Vs[4 * 512];  // [nt4][lane][16]
  __shared__ alignas(16) bf16_t Ps[4 * 512];  // [mt4][lane][16]
  __shared__ float Ss[64 * 33];
  __shared__ float red[64 * 4];
  __shared__ float mrow[64], lrow[64], arow[64];
  const int tid = threadIdx.x, lane = tid & 31, wid = tid >> 5;
  const int mw = wid & 3, nw = wid >> 2;
  const int bh = blockIdx.x, q0 = blockIdx.y * 64;
  const int b = bh >> 3, h = bh & 7;
  const size_t base = (size_t)bh << 16;
  // stage Q (A-fragment swizzle)
#pragma unroll
  for (int i = 0; i < 2; ++i) {
    int G = i * 256 + tid;
    int r = G >> 3, k = (G & 7) * 8, kc = k & 31;
    bf16_t* dst = Qs + ((k >> 5) * 4 + (r >> 4)) * 512 +
                  (((r & 15) + 16 * ((kc >> 3) & 1)) << 4) + 8 * (kc >> 4);
    cp16B(dst, Q + base + (size_t)(q0 + r) * 64 + k);
  }
  if (tid < 64) { mrow[tid] = -1e30f; lrow[tid] = 0.f; }
  f32x8 o[2] = {};
  const float scale = 0.125f;
  const int srow = tid >> 2, sq = tid & 3;
  for (int k0 = 0; k0 < Cc; k0 += 32) {
    {  // stage K^T (B-frag: contraction d contiguous in memory)
      int kk = tid >> 3, d = (tid & 7) * 8, dc = d & 31;
      bf16_t* dst = Ks + ((d >> 5) * 2 + (kk >> 4)) * 512 +
                  (((kk & 15) + 16 * ((dc >> 3) & 1)) << 4) + 8 * (dc >> 4);
      cp16B(dst, Kb + base + (size_t)(k0 + kk) * 64 + d);
    }
    {  // stage V (B-frag: contraction kk contiguous in transposed V)
      int d = tid >> 2, kk = (tid & 3) * 8;
      bf16_t* dst = Vs + (d >> 4) * 512 +
                    (((d & 15) + 16 * ((kk >> 3) & 1)) << 4) + 8 * (kk >> 4);
      cp16B(dst, Vt + base + (size_t)d * 1024 + k0 + kk);
    }
    __syncthreads();
    // S = Q @ K^T (each wave one 16x16 tile, K=64 via 2 WMMAs)
    f32x8 s = {};
    s = wmma_bf16(frag_ld(Qs + mw * 512), frag_ld(Ks + nw * 512), s);
    s = wmma_bf16(frag_ld(Qs + (4 + mw) * 512), frag_ld(Ks + (2 + nw) * 512), s);
    {
      int g = lane >> 4, n = lane & 15;
#pragma unroll
      for (int r = 0; r < 8; ++r) {
        int row = 16 * mw + 8 * g + r, col = 16 * nw + n;
        float v = s[r] * scale;
        if (mask[b * Cc + k0 + col] == 0) v = -1e10f;
        Ss[row * 33 + col] = v;
      }
    }
    __syncthreads();
    {  // partial max
      float mx = -1e30f;
#pragma unroll
      for (int j = 0; j < 8; ++j) mx = fmaxf(mx, Ss[srow * 33 + sq * 8 + j]);
      red[srow * 4 + sq] = mx;
    }
    __syncthreads();
    if (sq == 0) {
      float mx = fmaxf(fmaxf(red[srow * 4], red[srow * 4 + 1]),
                       fmaxf(red[srow * 4 + 2], red[srow * 4 + 3]));
      float mo = mrow[srow], mn = fmaxf(mo, mx);
      arow[srow] = __expf(mo - mn);
      mrow[srow] = mn;
    }
    __syncthreads();
    {  // exp + write P directly in A-fragment layout (one b128 store)
      float mn = mrow[srow], ps = 0.f, pv[8];
#pragma unroll
      for (int j = 0; j < 8; ++j) {
        pv[j] = __expf(Ss[srow * 33 + sq * 8 + j] - mn);
        ps += pv[j];
      }
      red[srow * 4 + sq] = ps;
      bf16_t* dst = Ps + (srow >> 4) * 512 +
                    (((srow & 15) + 16 * (sq & 1)) << 4) + 8 * (sq >> 1);
      store8_bf16(dst, pv);
    }
    __syncthreads();
    if (sq == 0)
      lrow[srow] = lrow[srow] * arow[srow] + red[srow * 4] + red[srow * 4 + 1] +
                   red[srow * 4 + 2] + red[srow * 4 + 3];
    {  // rescale running O
      int g = lane >> 4;
#pragma unroll
      for (int r = 0; r < 8; ++r) {
        float al = arow[16 * mw + 8 * g + r];
        o[0][r] *= al;
        o[1][r] *= al;
      }
    }
    // O += P @ V
    bf16x16 pa = frag_ld(Ps + mw * 512);
    o[0] = wmma_bf16(pa, frag_ld(Vs + (2 * nw) * 512), o[0]);
    o[1] = wmma_bf16(pa, frag_ld(Vs + (2 * nw + 1) * 512), o[1]);
    __syncthreads();
  }
  {
    int g = lane >> 4, n = lane & 15;
#pragma unroll
    for (int r = 0; r < 8; ++r) {
      int row = 16 * mw + 8 * g + r;
      float inv = 1.0f / lrow[row];
      int qrow = q0 + row;
#pragma unroll
      for (int j = 0; j < 2; ++j) {
        int col = 32 * nw + 16 * j + n;
        ctx[((size_t)(b * Cc + qrow)) * 512 + h * 64 + col] =
            (bf16_t)(o[j][r] * inv);
      }
    }
  }
}

// ---- residual + LayerNorm ---------------------------------------------------
__global__ __launch_bounds__(256) void ln_add(const float* __restrict__ x,
                                              const float* __restrict__ yv,
                                              const float* __restrict__ gg,
                                              const float* __restrict__ bb,
                                              float* __restrict__ of,
                                              bf16_t* __restrict__ ob) {
  __shared__ float rs[256], rq[256];
  const int t = blockIdx.x, tid = threadIdx.x;
  float v0 = x[(size_t)t * 512 + tid] + yv[(size_t)t * 512 + tid];
  float v1 = x[(size_t)t * 512 + 256 + tid] + yv[(size_t)t * 512 + 256 + tid];
  rs[tid] = v0 + v1;
  rq[tid] = v0 * v0 + v1 * v1;
  __syncthreads();
  for (int s = 128; s > 0; s >>= 1) {
    if (tid < s) { rs[tid] += rs[tid + s]; rq[tid] += rq[tid + s]; }
    __syncthreads();
  }
  float mean = rs[0] * (1.f / 512.f);
  float var = rq[0] * (1.f / 512.f) - mean * mean;
  float inv = rsqrtf(var + EPSf);
  float o0 = (v0 - mean) * inv * gg[tid] + bb[tid];
  float o1 = (v1 - mean) * inv * gg[256 + tid] + bb[256 + tid];
  of[(size_t)t * 512 + tid] = o0;
  of[(size_t)t * 512 + 256 + tid] = o1;
  ob[(size_t)t * 512 + tid] = (bf16_t)o0;
  ob[(size_t)t * 512 + 256 + tid] = (bf16_t)o1;
}

// ---- router + top2 ----------------------------------------------------------
__global__ __launch_bounds__(256) void router_topk(const float* __restrict__ src,
                                                   const float* __restrict__ rw,
                                                   int* __restrict__ sexp,
                                                   float* __restrict__ swt) {
  __shared__ float rwl[512 * 8];
  const int tid = threadIdx.x;
  for (int i = tid; i < 4096; i += 256) rwl[i] = rw[i];
  __syncthreads();
  const int t = blockIdx.x * 256 + tid;
  float lg[8] = {};
  const float* xr = src + (size_t)t * 512;
  for (int i = 0; i < 512; ++i) {
    float xv = xr[i];
#pragma unroll
    for (int e = 0; e < 8; ++e) lg[e] += xv * rwl[i * 8 + e];
  }
  float mx = lg[0];
#pragma unroll
  for (int e = 1; e < 8; ++e) mx = fmaxf(mx, lg[e]);
  float p[8];
#pragma unroll
  for (int e = 0; e < 8; ++e) p[e] = __expf(lg[e] - mx);
  int i0 = 0;
  float v0 = p[0];
#pragma unroll
  for (int e = 1; e < 8; ++e)
    if (p[e] > v0) { v0 = p[e]; i0 = e; }
  int i1 = (i0 == 0) ? 1 : 0;
  float v1 = p[i1];
#pragma unroll
  for (int e = 0; e < 8; ++e)
    if (e != i0 && p[e] > v1) { v1 = p[e]; i1 = e; }
  float tw = v0 + v1;
  sexp[2 * t] = i0;
  swt[2 * t] = v0 / tw;
  sexp[2 * t + 1] = i1;
  swt[2 * t + 1] = v1 / tw;
}

// ---- deterministic capacity assignment --------------------------------------
__global__ __launch_bounds__(256) void count_experts(const int* __restrict__ sexp,
                                                     int* __restrict__ counts) {
  __shared__ int cnt[8];
  const int tid = threadIdx.x;
  if (tid < 8) cnt[tid] = 0;
  __syncthreads();
  atomicAdd(&cnt[sexp[blockIdx.x * 256 + tid]], 1);
  __syncthreads();
  if (tid < 8) counts[blockIdx.x * 8 + tid] = cnt[tid];
}
__global__ void scan_blocks(const int* __restrict__ counts,
                            int* __restrict__ base) {
  int e = threadIdx.x;
  if (e < 8) {
    int run = 0;
    for (int b = 0; b < 128; ++b) {
      base[b * 8 + e] = run;
      run += counts[b * 8 + e];
    }
  }
}
__global__ __launch_bounds__(256) void assign_pos(const int* __restrict__ sexp,
                                                  const int* __restrict__ base,
                                                  int* __restrict__ spos) {
  __shared__ int wcnt[8][8];
  const int tid = threadIdx.x, lane = tid & 31, wid = tid >> 5;
  const int s = blockIdx.x * 256 + tid;
  const int e = sexp[s];
  int rank = 0;
  const unsigned lt = (1u << lane) - 1;
#pragma unroll
  for (int e0 = 0; e0 < 8; ++e0) {
    unsigned b32 = (unsigned)__ballot(e == e0);  // wave32: low 32 bits valid
    if (e == e0) rank = __popc(b32 & lt);
    if (lane == e0) wcnt[wid][e0] = __popc(b32);
  }
  __syncthreads();
  int off = base[blockIdx.x * 8 + e];
  for (int w = 0; w < wid; ++w) off += wcnt[w][e];
  spos[s] = off + rank;
}

// ---- dispatch scatter -------------------------------------------------------
__global__ __launch_bounds__(128) void scatter_disp(const bf16_t* __restrict__ sb,
                                                    const int* __restrict__ sexp,
                                                    const int* __restrict__ spos,
                                                    bf16_t* __restrict__ disp) {
  const int s = blockIdx.x;
  const int pos = spos[s];
  if (pos >= CAPc) return;
  const int e = sexp[s], t = s >> 1;
  const uint2* sr = (const uint2*)(sb + (size_t)t * 512);
  uint2* dr = (uint2*)(disp + ((size_t)e * CAPc + pos) * 512);
  dr[threadIdx.x] = sr[threadIdx.x];
}

// ---- fused expert FFN: y = gelu(X@w1+b1)@w2+b2  (grid 160 x 8) --------------
// w1t = bf16 [E][FF][D] (transposed), w2t = bf16 [E][D][FF] (transposed)
__global__ __launch_bounds__(256) void expert_ffn(const bf16_t* __restrict__ disp,
                                                  const bf16_t* __restrict__ w1t,
                                                  const float* __restrict__ b1,
                                                  const bf16_t* __restrict__ w2t,
                                                  const float* __restrict__ b2,
                                                  float* __restrict__ y) {
  extern __shared__ char smem[];
  bf16_t* Xs = (bf16_t*)smem;            // [ck16][mt2][lane][16] = 32768B
  bf16_t* W1c = (bf16_t*)(smem + 32768); // [ck2][nt2][lane][16]  =  4096B
  bf16_t* Hs = (bf16_t*)(smem + 36864);  // [mt2][lane][16]       =  2048B
  bf16_t* W2c = (bf16_t*)(smem + 38912); // [nt32][lane][16]      = 32768B
  float* Hred = (float*)(smem + 71680);  // 4*8*32 f32            =  4096B
  const int tid = threadIdx.x, lane = tid & 31, wid = tid >> 5;
  const int e = blockIdx.y, r0 = blockIdx.x * 32;
  const int tw = wid & 3, m1 = tw & 1, n1 = tw >> 1, hf = wid >> 2;
  const int g = lane >> 4, nn = lane & 15;
  const bf16_t* Xg = disp + ((size_t)e * CAPc + r0) * 512;
  // stage X once (A-fragment swizzle over 16 K-chunks)
#pragma unroll
  for (int i = 0; i < 8; ++i) {
    int G = i * 256 + tid;
    int r = G >> 6, k = (G & 63) * 8, kc = k & 31;
    bf16_t* dst = Xs + ((k >> 5) * 2 + (r >> 4)) * 512 +
                  (((r & 15) + 16 * ((kc >> 3) & 1)) << 4) + 8 * (kc >> 4);
    cp16B(dst, Xg + (size_t)r * 512 + k);
  }
  f32x8 acc[2][4] = {};
  for (int fc = 0; fc < FFf / 32; ++fc) {
    f32x8 hacc = {};
    for (int s8 = 0; s8 < 8; ++s8) {
      {  // stage W1 slab (contraction d contiguous in w1t)
        int n = tid & 31, d = (tid >> 5) * 8, dc = d & 31;
        bf16_t* dst = W1c + ((d >> 5) * 2 + (n >> 4)) * 512 +
                      (((n & 15) + 16 * ((dc >> 3) & 1)) << 4) + 8 * (dc >> 4);
        cp16B(dst, w1t + ((size_t)(e * FFf + fc * 32 + n)) * 512 + 64 * s8 + d);
      }
      __syncthreads();
      bf16x16 a = frag_ld(Xs + ((2 * s8 + hf) * 2 + m1) * 512);
      bf16x16 bb = frag_ld(W1c + (hf * 2 + n1) * 512);
      hacc = wmma_bf16(a, bb, hacc);
      __syncthreads();
    }
    if (hf == 1) {
#pragma unroll
      for (int r = 0; r < 8; ++r) Hred[(tw * 8 + r) * 32 + lane] = hacc[r];
    }
    __syncthreads();
    if (hf == 0) {
      const int col = 16 * n1 + nn;
      const int gA = (col >> 3) & 1;
      const int pA = 8 * (col >> 4) + (col & 7);
#pragma unroll
      for (int r = 0; r < 8; ++r) {
        float v = hacc[r] + Hred[(tw * 8 + r) * 32 + lane] +
                  b1[e * FFf + fc * 32 + col];
        float u = 0.7978845608028654f * (v + 0.044715f * v * v * v);
        float gl = 0.5f * v * (1.f + tanhf(u));
        // A-fragment swizzled store for H: row = 16*m1 + 8*g + r
        Hs[m1 * 512 + (((8 * g + r) + 16 * gA) << 4) + pA] = (bf16_t)gl;
      }
    }
    __syncthreads();
#pragma unroll
    for (int i = 0; i < 8; ++i) {  // stage W2 slab (contraction ff contiguous)
      int G = i * 256 + tid;
      int n = G >> 2, f = (G & 3) * 8;
      bf16_t* dst = W2c + (n >> 4) * 512 +
                    (((n & 15) + 16 * ((f >> 3) & 1)) << 4) + 8 * (f >> 4);
      cp16B(dst, w2t + ((size_t)(e * Dd + n)) * 2048 + fc * 32 + f);
    }
    __syncthreads();
#pragma unroll
    for (int m = 0; m < 2; ++m) {
      bf16x16 a = frag_ld(Hs + m * 512);
#pragma unroll
      for (int n = 0; n < 4; ++n) {
        bf16x16 bb = frag_ld(W2c + (4 * wid + n) * 512);
        acc[m][n] = wmma_bf16(a, bb, acc[m][n]);
      }
    }
    __syncthreads();
  }
#pragma unroll
  for (int m = 0; m < 2; ++m)
#pragma unroll
    for (int n = 0; n < 4; ++n)
#pragma unroll
      for (int r = 0; r < 8; ++r) {
        int row = r0 + 16 * m + 8 * g + r;
        int col = 64 * wid + 16 * n + nn;
        y[((size_t)e * CAPc + row) * 512 + col] = acc[m][n][r] + b2[e * 512 + col];
      }
}

// ---- combine expert outputs + residual + LN2 -> final output ----------------
__global__ __launch_bounds__(256) void combine_ln2(
    const float* __restrict__ src, const float* __restrict__ ym,
    const int* __restrict__ sexp, const int* __restrict__ spos,
    const float* __restrict__ sw, const float* __restrict__ gg,
    const float* __restrict__ bb, float* __restrict__ out) {
  __shared__ float rs[256], rq[256];
  const int t = blockIdx.x, tid = threadIdx.x;
  float v0 = src[(size_t)t * 512 + tid];
  float v1 = src[(size_t)t * 512 + 256 + tid];
  const int s0 = 2 * t, s1 = 2 * t + 1;
  int p0 = spos[s0], p1 = spos[s1];
  if (p0 < CAPc) {
    const float* yr = ym + ((size_t)sexp[s0] * CAPc + p0) * 512;
    float w = sw[s0];
    v0 += w * yr[tid];
    v1 += w * yr[256 + tid];
  }
  if (p1 < CAPc) {
    const float* yr = ym + ((size_t)sexp[s1] * CAPc + p1) * 512;
    float w = sw[s1];
    v0 += w * yr[tid];
    v1 += w * yr[256 + tid];
  }
  rs[tid] = v0 + v1;
  rq[tid] = v0 * v0 + v1 * v1;
  __syncthreads();
  for (int s = 128; s > 0; s >>= 1) {
    if (tid < s) { rs[tid] += rs[tid + s]; rq[tid] += rq[tid + s]; }
    __syncthreads();
  }
  float mean = rs[0] * (1.f / 512.f);
  float var = rq[0] * (1.f / 512.f) - mean * mean;
  float inv = rsqrtf(var + EPSf);
  out[(size_t)t * 512 + tid] = (v0 - mean) * inv * gg[tid] + bb[tid];
  out[(size_t)t * 512 + 256 + tid] =
      (v1 - mean) * inv * gg[256 + tid] + bb[256 + tid];
}

// ---- host side --------------------------------------------------------------
extern "C" void kernel_launch(void* const* d_in, const int* in_sizes, int n_in,
                              void* d_out, int out_size, void* d_ws,
                              size_t ws_size, hipStream_t stream) {
  (void)in_sizes; (void)n_in; (void)out_size; (void)ws_size;
  const float* x = (const float*)d_in[0];
  const int* src_mask = (const int*)d_in[1];
  const float* wq = (const float*)d_in[2];
  const float* bq = (const float*)d_in[3];
  const float* wk = (const float*)d_in[4];
  const float* bk = (const float*)d_in[5];
  const float* wv = (const float*)d_in[6];
  const float* bv = (const float*)d_in[7];
  const float* wo = (const float*)d_in[8];
  const float* bo = (const float*)d_in[9];
  const float* ln1_g = (const float*)d_in[10];
  const float* ln1_b = (const float*)d_in[11];
  const float* ln2_g = (const float*)d_in[12];
  const float* ln2_b = (const float*)d_in[13];
  const float* router_w = (const float*)d_in[14];
  const float* w1 = (const float*)d_in[15];
  const float* b1 = (const float*)d_in[16];
  const float* w2 = (const float*)d_in[17];
  const float* b2 = (const float*)d_in[18];
  char* ws = (char*)d_ws;

  // workspace layout (bytes); phase-B buffers alias dead phase-A buffers
  const size_t OFF_Q = 0, OFF_K = 16777216, OFF_V = 33554432;
  const size_t OFF_CTX = 50331648, OFF_Y1 = 67108864;
  const size_t OFF_DISP = 0, OFF_YMOE = 41943040;  // alias q/k/v/ctx/y1 region
  const size_t OFF_SRC = 125829120, OFF_SRCB = 159383552;
  const size_t OFF_W1T = 176160768, OFF_W2T = 192937984;
  const size_t OFF_WQT = 209715200, OFF_WKT = 210239488;
  const size_t OFF_WVT = 210763776, OFF_WOT = 211288064;
  const size_t OFF_SEXP = 211812352, OFF_SPOS = OFF_SEXP + 131072;
  const size_t OFF_SW = OFF_SPOS + 131072, OFF_CNT = OFF_SW + 131072;
  const size_t OFF_BASE = OFF_CNT + 4096;

  bf16_t* qb = (bf16_t*)(ws + OFF_Q);
  bf16_t* kb = (bf16_t*)(ws + OFF_K);
  bf16_t* vt = (bf16_t*)(ws + OFF_V);
  bf16_t* ctx = (bf16_t*)(ws + OFF_CTX);
  float* y1 = (float*)(ws + OFF_Y1);
  bf16_t* disp = (bf16_t*)(ws + OFF_DISP);
  float* ymoe = (float*)(ws + OFF_YMOE);
  float* srcf = (float*)(ws + OFF_SRC);
  bf16_t* srcb = (bf16_t*)(ws + OFF_SRCB);
  bf16_t* w1t = (bf16_t*)(ws + OFF_W1T);
  bf16_t* w2t = (bf16_t*)(ws + OFF_W2T);
  bf16_t* wqt = (bf16_t*)(ws + OFF_WQT);
  bf16_t* wkt = (bf16_t*)(ws + OFF_WKT);
  bf16_t* wvt = (bf16_t*)(ws + OFF_WVT);
  bf16_t* wot = (bf16_t*)(ws + OFF_WOT);
  int* sexp = (int*)(ws + OFF_SEXP);
  int* spos = (int*)(ws + OFF_SPOS);
  float* swt = (float*)(ws + OFF_SW);
  int* cnts = (int*)(ws + OFF_CNT);
  int* base = (int*)(ws + OFF_BASE);

  // transpose+convert all weights to bf16 [N][K] layouts (one-time, L2 fits)
  transpose_f2b<<<dim3(16, 16, 1), 256, 0, stream>>>(wq, wqt, 512, 512);
  transpose_f2b<<<dim3(16, 16, 1), 256, 0, stream>>>(wk, wkt, 512, 512);
  transpose_f2b<<<dim3(16, 16, 1), 256, 0, stream>>>(wv, wvt, 512, 512);
  transpose_f2b<<<dim3(16, 16, 1), 256, 0, stream>>>(wo, wot, 512, 512);
  transpose_f2b<<<dim3(64, 16, 8), 256, 0, stream>>>(w1, w1t, 512, 2048);
  transpose_f2b<<<dim3(16, 64, 8), 256, 0, stream>>>(w2, w2t, 2048, 512);

  dim3 gg(Tt / 128, Dd / 64);
  gemm_wmma<false, 1><<<gg, 256, 0, stream>>>(x, wqt, bq, qb, Tt, Dd, Dd);
  gemm_wmma<false, 1><<<gg, 256, 0, stream>>>(x, wkt, bk, kb, Tt, Dd, Dd);
  gemm_wmma<false, 2><<<gg, 256, 0, stream>>>(x, wvt, bv, vt, Tt, Dd, Dd);

  flash_attn<<<dim3(Bb * Hh, Cc / 64), 256, 0, stream>>>(qb, kb, vt, src_mask,
                                                         ctx);

  gemm_wmma<true, 0><<<gg, 256, 0, stream>>>(ctx, wot, bo, y1, Tt, Dd, Dd);

  ln_add<<<Tt, 256, 0, stream>>>(x, y1, ln1_g, ln1_b, srcf, srcb);

  router_topk<<<Tt / 256, 256, 0, stream>>>(srcf, router_w, sexp, swt);
  count_experts<<<128, 256, 0, stream>>>(sexp, cnts);
  scan_blocks<<<1, 64, 0, stream>>>(cnts, base);
  assign_pos<<<128, 256, 0, stream>>>(sexp, base, spos);
  scatter_disp<<<Tt * 2, 128, 0, stream>>>(srcb, sexp, spos, disp);

  expert_ffn<<<dim3(CAPc / 32, Ee), 256, 75776, stream>>>(disp, w1t, b1, w2t, b2,
                                                          ymoe);

  combine_ln2<<<Tt, 256, 0, stream>>>(srcf, ymoe, sexp, spos, swt, ln2_g, ln2_b,
                                      (float*)d_out);
}
